// TransformerBlock_24464133718701
// MI455X (gfx1250) — matrix operand order repaired and checked by reading, not compiled, not run on hardware
//
#include <hip/hip_runtime.h>

// ---------------- problem constants ----------------
#define B_SZ   2
#define SEQL   2048
#define DM     1024
#define NH     16
#define DK     64
#define DFF    4096
#define MTOK   (B_SZ * SEQL)   // 4096 rows of activations

typedef _Float16 half_t;
typedef __attribute__((ext_vector_type(16))) _Float16 v16h;
typedef __attribute__((ext_vector_type(8)))  _Float16 v8h;
typedef __attribute__((ext_vector_type(8)))  float    v8f;

static __device__ __forceinline__ v16h mkv16(v8h lo, v8h hi) {
  v16h r;
#pragma unroll
  for (int i = 0; i < 8; ++i) { r[i] = lo[i]; r[i + 8] = hi[i]; }
  return r;
}

static __device__ __forceinline__ v8f wmma16(v16h a, v16h b, v8f c) {
  // D = A(16x32,f16) * B(32x16,f16) + C(16x16,f32)
  return __builtin_amdgcn_wmma_f32_16x16x32_f16(false, a, false, b, (short)0, c,
                                                false, false);
}

// CDNA5 async global->LDS DMA (16B per lane), tracked by ASYNCcnt.
// lds = byte offset of destination within the workgroup LDS allocation
// (low 32 bits of the generic pointer to a __shared__ object).
static __device__ __forceinline__ void async_ld_b128(unsigned lds,
                                                     const void* gaddr) {
  asm volatile("global_load_async_to_lds_b128 %0, %1, off"
               :: "v"(lds), "v"(gaddr) : "memory");
}
template <int N>
static __device__ __forceinline__ void wait_async() {
  asm volatile("s_wait_asynccnt %0" :: "n"(N) : "memory");
}
static __device__ __forceinline__ unsigned lds_off(const void* p) {
  return (unsigned)(size_t)p;   // LDS aperture: addr[31:0] == LDS byte offset
}

// ---------------- small utility kernels ----------------

__global__ void cvt_f32_to_f16(const float* __restrict__ in,
                               half_t* __restrict__ out, int n4) {
  int i = blockIdx.x * blockDim.x + threadIdx.x;
  if (i < n4) {
    float4 v = ((const float4*)in)[i];
    out[i * 4 + 0] = (half_t)v.x;
    out[i * 4 + 1] = (half_t)v.y;
    out[i * 4 + 2] = (half_t)v.z;
    out[i * 4 + 3] = (half_t)v.w;
  }
}

__global__ void rope_tables(float* __restrict__ Ct, float* __restrict__ St) {
  int t = blockIdx.x * blockDim.x + threadIdx.x;
  if (t < SEQL * 32) {
    int s = t >> 5, i = t & 31;
    float freq = __expf(-logf(10000.0f) * (float)i / 32.0f);
    float ang = (float)s * freq;
    Ct[t] = cosf(ang);
    St[t] = sinf(ang);
  }
}

// one block per row; 256 threads x 4 cols = 1024
__global__ __launch_bounds__(256) void rmsnorm_f16(
    const float* __restrict__ X, const float* __restrict__ W,
    half_t* __restrict__ Out) {
  __shared__ float red[256];
  const int row = blockIdx.x;
  const float* xr = X + (size_t)row * DM;
  float4 xv = ((const float4*)xr)[threadIdx.x];
  red[threadIdx.x] = xv.x * xv.x + xv.y * xv.y + xv.z * xv.z + xv.w * xv.w;
  __syncthreads();
  for (int s = 128; s > 0; s >>= 1) {
    if (threadIdx.x < s) red[threadIdx.x] += red[threadIdx.x + s];
    __syncthreads();
  }
  float scale = rsqrtf(red[0] / (float)DM + 1e-5f);
  float4 wv = ((const float4*)W)[threadIdx.x];
  size_t o = (size_t)row * DM + threadIdx.x * 4;
  Out[o + 0] = (half_t)(xv.x * scale * wv.x);
  Out[o + 1] = (half_t)(xv.y * scale * wv.y);
  Out[o + 2] = (half_t)(xv.z * scale * wv.z);
  Out[o + 3] = (half_t)(xv.w * scale * wv.w);
}

// ---------------- generic WMMA GEMM:  C[M][N] = A[M][K] * W[N][K]^T -------
// Double-buffered LDS, global->LDS via async DMA (ASYNCcnt pipeline).
enum { EPI_RES_F32 = 0, EPI_ROPE_Q = 2, EPI_ROPE_K = 3, EPI_V_BHSD = 4 };

template <int EPI>
__global__ __launch_bounds__(256) void gemm_f16(
    const half_t* __restrict__ A, const half_t* __restrict__ W,
    const float* __restrict__ Res, float* __restrict__ OutF,
    half_t* __restrict__ OutH, const float* __restrict__ Ct,
    const float* __restrict__ St, int M, int N, int K) {
  __shared__ __align__(16) half_t Ash[2][128][40];
  __shared__ __align__(16) half_t Bsh[2][128][40];
  const int tid = threadIdx.x;
  const int lane = tid & 31, wid = tid >> 5;
  const int wm = wid & 3, wn = wid >> 2;       // 4 waves in M, 2 in N
  const int hl = lane >> 4, l15 = lane & 15;   // lane half, lane-in-half
  const int m0 = blockIdx.y * 128, n0 = blockIdx.x * 128;

  // loader mapping: 512 chunks of 8 halves; this thread owns chunks tid, tid+256
  const int r0 = tid >> 2, s0 = tid & 3;       // rows 0..63 / 64..127, seg 0..3
  const half_t* gA0 = A + (size_t)(m0 + r0) * K + s0 * 8;
  const half_t* gA1 = gA0 + (size_t)64 * K;
  const half_t* gB0 = W + (size_t)(n0 + r0) * K + s0 * 8;
  const half_t* gB1 = gB0 + (size_t)64 * K;
  unsigned lA0[2], lA1[2], lB0[2], lB1[2];
#pragma unroll
  for (int b = 0; b < 2; ++b) {
    lA0[b] = lds_off(&Ash[b][r0][s0 * 8]);
    lA1[b] = lds_off(&Ash[b][r0 + 64][s0 * 8]);
    lB0[b] = lds_off(&Bsh[b][r0][s0 * 8]);
    lB1[b] = lds_off(&Bsh[b][r0 + 64][s0 * 8]);
  }

  v8f c[2][4];
#pragma unroll
  for (int mt = 0; mt < 2; ++mt)
#pragma unroll
    for (int nt = 0; nt < 4; ++nt) c[mt][nt] = (v8f){0, 0, 0, 0, 0, 0, 0, 0};

  // prologue: stage tile 0 into buffer 0
  async_ld_b128(lA0[0], gA0);
  async_ld_b128(lA1[0], gA1);
  async_ld_b128(lB0[0], gB0);
  async_ld_b128(lB1[0], gB1);

  const int ntiles = K >> 5;
  for (int it = 0; it < ntiles; ++it) {
    const int buf = it & 1;
    if (it + 1 < ntiles) {           // stage next tile into the other buffer
      const int o = (it + 1) << 5;
      async_ld_b128(lA0[buf ^ 1], gA0 + o);
      async_ld_b128(lA1[buf ^ 1], gA1 + o);
      async_ld_b128(lB0[buf ^ 1], gB0 + o);
      async_ld_b128(lB1[buf ^ 1], gB1 + o);
      wait_async<4>();               // in-order: current tile resident
    } else {
      wait_async<0>();
    }
    __syncthreads();

    v16h a[2];
#pragma unroll
    for (int mt = 0; mt < 2; ++mt) {
      int r = wm * 32 + mt * 16 + l15;
      int kb = hl * 16;
      a[mt] = mkv16(*(const v8h*)&Ash[buf][r][kb],
                    *(const v8h*)&Ash[buf][r][kb + 8]);
    }
#pragma unroll
    for (int nt = 0; nt < 4; ++nt) {
      int n = wn * 64 + nt * 16 + l15;
      int kb = hl * 16;
      v16h b = mkv16(*(const v8h*)&Bsh[buf][n][kb],
                     *(const v8h*)&Bsh[buf][n][kb + 8]);
#pragma unroll
      for (int mt = 0; mt < 2; ++mt) c[mt][nt] = wmma16(a[mt], b, c[mt][nt]);
    }
    __syncthreads();   // all reads of this buffer done before it is re-filled
  }

#pragma unroll
  for (int mt = 0; mt < 2; ++mt)
#pragma unroll
    for (int nt = 0; nt < 4; ++nt)
#pragma unroll
      for (int v = 0; v < 8; ++v) {
        int m = m0 + wm * 32 + mt * 16 + hl * 8 + v;
        int n = n0 + wn * 64 + nt * 16 + l15;
        float val = c[mt][nt][v];
        if constexpr (EPI == EPI_RES_F32) {
          OutF[(size_t)m * N + n] = val + Res[(size_t)m * N + n];
        } else if constexpr (EPI == EPI_V_BHSD) {
          int b = m >> 11, s = m & (SEQL - 1), h = n >> 6, d = n & 63;
          OutH[(((size_t)(b * NH + h)) * SEQL + s) * DK + d] = (half_t)val;
        } else {  // RoPE (Q or K)
          float other = __shfl_xor(val, 1, 32);
          int b = m >> 11, s = m & (SEQL - 1), h = n >> 6, d = n & 63;
          int fi = d >> 1;
          float cs = Ct[s * 32 + fi], sn = St[s * 32 + fi];
          float r = ((n & 1) == 0) ? (val * cs - other * sn)
                                   : (other * sn + val * cs);
          if constexpr (EPI == EPI_ROPE_Q) r *= 0.125f;  // 1/sqrt(64)
          OutH[(((size_t)(b * NH + h)) * SEQL + s) * DK + d] = (half_t)r;
        }
      }
}

// ---------------- dual GEMM + SiLU*mul (SwiGLU up-projection) -------------
__global__ __launch_bounds__(256) void gemm_dual_silu(
    const half_t* __restrict__ A, const half_t* __restrict__ W1,
    const half_t* __restrict__ W3, half_t* __restrict__ OutH, int M, int N,
    int K) {
  __shared__ __align__(16) half_t Ash[2][128][40];
  __shared__ __align__(16) half_t B1[2][128][40];
  __shared__ __align__(16) half_t B2[2][128][40];
  const int tid = threadIdx.x;
  const int lane = tid & 31, wid = tid >> 5;
  const int wm = wid & 3, wn = wid >> 2;
  const int hl = lane >> 4, l15 = lane & 15;
  const int m0 = blockIdx.y * 128, n0 = blockIdx.x * 128;

  const int r0 = tid >> 2, s0 = tid & 3;
  const half_t* gA0 = A + (size_t)(m0 + r0) * K + s0 * 8;
  const half_t* gA1 = gA0 + (size_t)64 * K;
  const half_t* g10 = W1 + (size_t)(n0 + r0) * K + s0 * 8;
  const half_t* g11 = g10 + (size_t)64 * K;
  const half_t* g30 = W3 + (size_t)(n0 + r0) * K + s0 * 8;
  const half_t* g31 = g30 + (size_t)64 * K;
  unsigned lA0[2], lA1[2], l10[2], l11[2], l30[2], l31[2];
#pragma unroll
  for (int b = 0; b < 2; ++b) {
    lA0[b] = lds_off(&Ash[b][r0][s0 * 8]);
    lA1[b] = lds_off(&Ash[b][r0 + 64][s0 * 8]);
    l10[b] = lds_off(&B1[b][r0][s0 * 8]);
    l11[b] = lds_off(&B1[b][r0 + 64][s0 * 8]);
    l30[b] = lds_off(&B2[b][r0][s0 * 8]);
    l31[b] = lds_off(&B2[b][r0 + 64][s0 * 8]);
  }

  v8f c1[2][4], c2[2][4];
#pragma unroll
  for (int mt = 0; mt < 2; ++mt)
#pragma unroll
    for (int nt = 0; nt < 4; ++nt) {
      c1[mt][nt] = (v8f){0, 0, 0, 0, 0, 0, 0, 0};
      c2[mt][nt] = (v8f){0, 0, 0, 0, 0, 0, 0, 0};
    }

  async_ld_b128(lA0[0], gA0);
  async_ld_b128(lA1[0], gA1);
  async_ld_b128(l10[0], g10);
  async_ld_b128(l11[0], g11);
  async_ld_b128(l30[0], g30);
  async_ld_b128(l31[0], g31);

  const int ntiles = K >> 5;
  for (int it = 0; it < ntiles; ++it) {
    const int buf = it & 1;
    if (it + 1 < ntiles) {
      const int o = (it + 1) << 5;
      async_ld_b128(lA0[buf ^ 1], gA0 + o);
      async_ld_b128(lA1[buf ^ 1], gA1 + o);
      async_ld_b128(l10[buf ^ 1], g10 + o);
      async_ld_b128(l11[buf ^ 1], g11 + o);
      async_ld_b128(l30[buf ^ 1], g30 + o);
      async_ld_b128(l31[buf ^ 1], g31 + o);
      wait_async<6>();
    } else {
      wait_async<0>();
    }
    __syncthreads();

    v16h a[2];
#pragma unroll
    for (int mt = 0; mt < 2; ++mt) {
      int r = wm * 32 + mt * 16 + l15;
      int kb = hl * 16;
      a[mt] = mkv16(*(const v8h*)&Ash[buf][r][kb],
                    *(const v8h*)&Ash[buf][r][kb + 8]);
    }
#pragma unroll
    for (int nt = 0; nt < 4; ++nt) {
      int n = wn * 64 + nt * 16 + l15;
      int kb = hl * 16;
      v16h b1 = mkv16(*(const v8h*)&B1[buf][n][kb],
                      *(const v8h*)&B1[buf][n][kb + 8]);
      v16h b2 = mkv16(*(const v8h*)&B2[buf][n][kb],
                      *(const v8h*)&B2[buf][n][kb + 8]);
#pragma unroll
      for (int mt = 0; mt < 2; ++mt) {
        c1[mt][nt] = wmma16(a[mt], b1, c1[mt][nt]);
        c2[mt][nt] = wmma16(a[mt], b2, c2[mt][nt]);
      }
    }
    __syncthreads();
  }

#pragma unroll
  for (int mt = 0; mt < 2; ++mt)
#pragma unroll
    for (int nt = 0; nt < 4; ++nt)
#pragma unroll
      for (int v = 0; v < 8; ++v) {
        int m = m0 + wm * 32 + mt * 16 + hl * 8 + v;
        int n = n0 + wn * 64 + nt * 16 + l15;
        float g = c1[mt][nt][v];
        float s = c2[mt][nt][v];
        float sig = 1.0f / (1.0f + __expf(-g));
        OutH[(size_t)m * N + n] = (half_t)(g * sig * s);
      }
}

// ---------------- flash attention (causal, per (b,h), 64 q-rows / WG) -----
__global__ __launch_bounds__(128) void flash_attn(
    const half_t* __restrict__ Q, const half_t* __restrict__ Kh,
    const half_t* __restrict__ V, half_t* __restrict__ O) {
  __shared__ __align__(16) half_t Qsh[64][72];
  __shared__ __align__(16) half_t Ksh[64][72];
  __shared__ __align__(16) half_t Vt[64][72];       // Vt[d][key]
  __shared__ __align__(16) half_t Psh[4][16][72];   // per-wave P tile
  const int tid = threadIdx.x, lane = tid & 31, wid = tid >> 5;
  const int hl = lane >> 4, l15 = lane & 15;
  const int bh = blockIdx.y;               // b*NH + h
  const int row0 = blockIdx.x * 64;
  const size_t base = (size_t)bh * SEQL * DK;

#pragma unroll
  for (int i = 0; i < 4; ++i) {            // stage Q block (64x64)
    int ch = tid + i * 128;
    int r = ch >> 3, seg = ch & 7;
    *(v8h*)&Qsh[r][seg * 8] =
        *(const v8h*)&Q[base + (size_t)(row0 + r) * DK + seg * 8];
  }
  __syncthreads();
  v16h aq[2];
  {
    int r = wid * 16 + l15;
    int kb = hl * 16;
    aq[0] = mkv16(*(const v8h*)&Qsh[r][kb], *(const v8h*)&Qsh[r][kb + 8]);
    aq[1] = mkv16(*(const v8h*)&Qsh[r][32 + kb], *(const v8h*)&Qsh[r][40 + kb]);
  }

  v8f o[4];
  float m_i[8], l_i[8];
#pragma unroll
  for (int nt = 0; nt < 4; ++nt) o[nt] = (v8f){0, 0, 0, 0, 0, 0, 0, 0};
#pragma unroll
  for (int v = 0; v < 8; ++v) { m_i[v] = -1e30f; l_i[v] = 0.0f; }

  for (int j0 = 0; j0 < row0 + 64; j0 += 64) {
#pragma unroll
    for (int i = 0; i < 4; ++i) {          // stage K block + V transposed
      int ch = tid + i * 128;
      int r = ch >> 3, seg = ch & 7;
      *(v8h*)&Ksh[r][seg * 8] =
          *(const v8h*)&Kh[base + (size_t)(j0 + r) * DK + seg * 8];
      v8h vv = *(const v8h*)&V[base + (size_t)(j0 + r) * DK + seg * 8];
#pragma unroll
      for (int e = 0; e < 8; ++e) Vt[seg * 8 + e][r] = vv[e];
    }
    __syncthreads();

    v8f s[4];
#pragma unroll
    for (int nt = 0; nt < 4; ++nt) s[nt] = (v8f){0, 0, 0, 0, 0, 0, 0, 0};
#pragma unroll
    for (int ks = 0; ks < 2; ++ks) {       // S = Q * K^T  (K dim = d = 64)
#pragma unroll
      for (int nt = 0; nt < 4; ++nt) {
        int n = nt * 16 + l15;
        int kb = ks * 32 + hl * 16;
        v16h bf = mkv16(*(const v8h*)&Ksh[n][kb], *(const v8h*)&Ksh[n][kb + 8]);
        s[nt] = wmma16(aq[ks], bf, s[nt]);
      }
    }

    // causal mask + online softmax (row stats live on 16-lane halves)
#pragma unroll
    for (int v = 0; v < 8; ++v) {
      int qi = row0 + wid * 16 + hl * 8 + v;
      float sv[4];
      float lmax = -1e30f;
#pragma unroll
      for (int nt = 0; nt < 4; ++nt) {
        int kj = j0 + nt * 16 + l15;
        float xx = s[nt][v];
        if (kj > qi) xx = -1e30f;
        sv[nt] = xx;
        lmax = fmaxf(lmax, xx);
      }
#pragma unroll
      for (int msk = 1; msk < 16; msk <<= 1)
        lmax = fmaxf(lmax, __shfl_xor(lmax, msk, 32));
      float mnew = fmaxf(m_i[v], lmax);
      float alpha = __expf(m_i[v] - mnew);
      m_i[v] = mnew;
      float lsum = 0.0f;
#pragma unroll
      for (int nt = 0; nt < 4; ++nt) {
        float p = __expf(sv[nt] - mnew);
        lsum += p;
        Psh[wid][hl * 8 + v][nt * 16 + l15] = (half_t)p;
      }
#pragma unroll
      for (int msk = 1; msk < 16; msk <<= 1) lsum += __shfl_xor(lsum, msk, 32);
      l_i[v] = l_i[v] * alpha + lsum;
#pragma unroll
      for (int nt = 0; nt < 4; ++nt) o[nt][v] *= alpha;
    }
    __syncthreads();

#pragma unroll
    for (int ks = 0; ks < 2; ++ks) {       // O += P * V   (K dim = keys = 64)
      int r = l15;
      int kb = ks * 32 + hl * 16;
      v16h ap =
          mkv16(*(const v8h*)&Psh[wid][r][kb], *(const v8h*)&Psh[wid][r][kb + 8]);
#pragma unroll
      for (int nt = 0; nt < 4; ++nt) {
        int n = nt * 16 + l15;
        int kb2 = ks * 32 + hl * 16;
        v16h bv = mkv16(*(const v8h*)&Vt[n][kb2], *(const v8h*)&Vt[n][kb2 + 8]);
        o[nt] = wmma16(ap, bv, o[nt]);
      }
    }
    __syncthreads();
  }

  const int bb = bh >> 4, hh2 = bh & 15;
#pragma unroll
  for (int v = 0; v < 8; ++v) {
    int srow = row0 + wid * 16 + hl * 8 + v;
    float inv = 1.0f / l_i[v];
#pragma unroll
    for (int nt = 0; nt < 4; ++nt) {
      int d = nt * 16 + l15;
      O[((size_t)(bb * SEQL + srow)) * DM + hh2 * DK + d] =
          (half_t)(o[nt][v] * inv);
    }
  }
}

// ---------------- host-side orchestration ----------------
extern "C" void kernel_launch(void* const* d_in, const int* in_sizes, int n_in,
                              void* d_out, int out_size, void* d_ws,
                              size_t ws_size, hipStream_t stream) {
  (void)in_sizes; (void)n_in; (void)out_size; (void)ws_size;
  const float* x   = (const float*)d_in[0];
  const float* wq  = (const float*)d_in[1];
  const float* wk  = (const float*)d_in[2];
  const float* wv  = (const float*)d_in[3];
  const float* wo  = (const float*)d_in[4];
  const float* ln1 = (const float*)d_in[5];
  const float* ln2 = (const float*)d_in[6];
  const float* w1  = (const float*)d_in[7];
  const float* w3  = (const float*)d_in[8];
  const float* w2  = (const float*)d_in[9];
  float* out = (float*)d_out;

  char* ws = (char*)d_ws;
  size_t off = 0;
  auto alloc = [&](size_t bytes) -> void* {
    void* p = ws + off;
    off = (off + bytes + 255) & ~(size_t)255;
    return p;
  };
  half_t* xnh   = (half_t*)alloc((size_t)MTOK * DM * 2);
  half_t* qh    = (half_t*)alloc((size_t)MTOK * DM * 2);
  half_t* kh    = (half_t*)alloc((size_t)MTOK * DM * 2);
  half_t* vh    = (half_t*)alloc((size_t)MTOK * DM * 2);
  half_t* atth  = (half_t*)alloc((size_t)MTOK * DM * 2);
  float*  x1    = (float*) alloc((size_t)MTOK * DM * 4);
  half_t* hh    = (half_t*)alloc((size_t)MTOK * DFF * 2);
  half_t* wqh   = (half_t*)alloc((size_t)DM * DM * 2);
  half_t* wkh   = (half_t*)alloc((size_t)DM * DM * 2);
  half_t* wvh   = (half_t*)alloc((size_t)DM * DM * 2);
  half_t* woh   = (half_t*)alloc((size_t)DM * DM * 2);
  half_t* w1h   = (half_t*)alloc((size_t)DFF * DM * 2);
  half_t* w3h   = (half_t*)alloc((size_t)DFF * DM * 2);
  half_t* w2h   = (half_t*)alloc((size_t)DM * DFF * 2);
  float*  Ct    = (float*) alloc((size_t)SEQL * 32 * 4);
  float*  St    = (float*) alloc((size_t)SEQL * 32 * 4);

  // weight conversion f32 -> f16
  auto cvt = [&](const float* src, half_t* dst, size_t n) {
    int n4 = (int)(n / 4);
    cvt_f32_to_f16<<<(n4 + 255) / 256, 256, 0, stream>>>(src, dst, n4);
  };
  cvt(wq, wqh, (size_t)DM * DM);
  cvt(wk, wkh, (size_t)DM * DM);
  cvt(wv, wvh, (size_t)DM * DM);
  cvt(wo, woh, (size_t)DM * DM);
  cvt(w1, w1h, (size_t)DFF * DM);
  cvt(w3, w3h, (size_t)DFF * DM);
  cvt(w2, w2h, (size_t)DM * DFF);

  rope_tables<<<(SEQL * 32 + 255) / 256, 256, 0, stream>>>(Ct, St);

  // ---- attention branch ----
  rmsnorm_f16<<<MTOK, 256, 0, stream>>>(x, ln1, xnh);

  dim3 gQKV(DM / 128, MTOK / 128);
  gemm_f16<EPI_ROPE_Q><<<gQKV, 256, 0, stream>>>(xnh, wqh, nullptr, nullptr,
                                                 qh, Ct, St, MTOK, DM, DM);
  gemm_f16<EPI_ROPE_K><<<gQKV, 256, 0, stream>>>(xnh, wkh, nullptr, nullptr,
                                                 kh, Ct, St, MTOK, DM, DM);
  gemm_f16<EPI_V_BHSD><<<gQKV, 256, 0, stream>>>(xnh, wvh, nullptr, nullptr,
                                                 vh, Ct, St, MTOK, DM, DM);

  flash_attn<<<dim3(SEQL / 64, B_SZ * NH), 128, 0, stream>>>(qh, kh, vh, atth);

  // x1 = x + attn_out * Wo^T
  gemm_f16<EPI_RES_F32><<<gQKV, 256, 0, stream>>>(atth, woh, x, x1, nullptr,
                                                  Ct, St, MTOK, DM, DM);

  // ---- FFN branch ----
  rmsnorm_f16<<<MTOK, 256, 0, stream>>>(x1, ln2, xnh);

  gemm_dual_silu<<<dim3(DFF / 128, MTOK / 128), 256, 0, stream>>>(
      xnh, w1h, w3h, hh, MTOK, DFF, DM);

  // out = x1 + h * W2^T
  gemm_f16<EPI_RES_F32><<<dim3(DM / 128, MTOK / 128), 256, 0, stream>>>(
      hh, w2h, x1, out, nullptr, Ct, St, MTOK, DM, DFF);
}
